// Graphformer_53652731462352
// MI455X (gfx1250) — compile-verified, matrix operand order checked by reading, not run on hardware
//
#include <hip/hip_runtime.h>
#include <hip/hip_bf16.h>

// ---------------------------------------------------------------------------
// CDNA5 / gfx1250 graph-transformer forward.  bf16 WMMA GEMMs (f32 accum),
// fused epilogues, scalar kernels for the per-edge gather/scatter.
// ---------------------------------------------------------------------------

typedef __bf16 v16bf __attribute__((ext_vector_type(16)));
typedef float  v8f   __attribute__((ext_vector_type(8)));

#define CDIM   128
#define HWC    16384
#define NNODE  32768
#define SC0f   0.625f      // (40-(-40))/128
#define SC1f   2.2f        // (140.8-(-140.8))/128

__device__ __forceinline__ __bf16 f2bf(float f) {
  unsigned u = __builtin_bit_cast(unsigned, f);
  unsigned r = u + 0x7FFFu + ((u >> 16) & 1u);     // round-to-nearest-even
  unsigned short h = (unsigned short)(r >> 16);
  return __builtin_bit_cast(__bf16, h);
}

// Load a 16x32 bf16 fragment (A layout; B uses the mirrored layout with the
// operand stored transposed [N x K], so the same loader serves both).
__device__ __forceinline__ v16bf load_frag(const __bf16* base, int ld) {
  int lane = threadIdx.x & 31;
  int lo = lane & 15, hi = lane >> 4;
  const __bf16* p = base + (long)lo * ld;
  union { unsigned u[8]; v16bf v; } f;
#pragma unroll
  for (int t = 0; t < 8; ++t) {
    int kb = (t < 4) ? (hi * 8 + 2 * t) : (16 + hi * 8 + 2 * (t - 4));
    f.u[t] = *reinterpret_cast<const unsigned*>(p + kb);
  }
  return f.v;
}

// D[M x Ncols] = A[M x K] * Bt[Ncols x K]^T.  One wave -> 16x128 strip.
// mode 0: store f32.  mode 1: bf16(relu(x + bias)).  mode 2: bf16(x).
__global__ __launch_bounds__(256) void wmma_gemm(
    const __bf16* __restrict__ A, int lda,
    const __bf16* __restrict__ Bt, int ldb,
    float* __restrict__ Df, __bf16* __restrict__ Dbf, int ldd,
    const float* __restrict__ bias, int K, int mode) {
  int wave = threadIdx.x >> 5;
  int row0 = blockIdx.x * 128 + wave * 16;
  int col0 = blockIdx.y * 128;
  v8f acc[8];
#pragma unroll
  for (int t = 0; t < 8; ++t)
#pragma unroll
    for (int r = 0; r < 8; ++r) acc[t][r] = 0.0f;

  for (int k0 = 0; k0 < K; k0 += 32) {
    v16bf a = load_frag(A + (long)row0 * lda + k0, lda);
#pragma unroll
    for (int t = 0; t < 8; ++t) {
      v16bf b = load_frag(Bt + (long)(col0 + t * 16) * ldb + k0, ldb);
      acc[t] = __builtin_amdgcn_wmma_f32_16x16x32_bf16(
          false, a, false, b, (short)0, acc[t], false, false);
    }
  }
  int lane = threadIdx.x & 31, lo = lane & 15, hi = lane >> 4;
#pragma unroll
  for (int t = 0; t < 8; ++t) {
    int col = col0 + t * 16 + lo;
#pragma unroll
    for (int r = 0; r < 8; ++r) {
      int row = row0 + hi * 8 + r;
      long idx = (long)row * ldd + col;
      float v = acc[t][r];
      if (mode == 0) {
        Df[idx] = v;
      } else if (mode == 1) {
        float x = v + bias[col];
        Dbf[idx] = f2bf(x > 0.0f ? x : 0.0f);
      } else {
        Dbf[idx] = f2bf(v);
      }
    }
  }
}

// ---- prep kernels ---------------------------------------------------------

__global__ void cvt_bf16(const float* __restrict__ in, __bf16* __restrict__ out, int n) {
  int i = blockIdx.x * blockDim.x + threadIdx.x;
  if (i < n) out[i] = f2bf(in[i]);
}

// W: R x Ccols row-major  ->  Wt[c*R + r] (bf16 transposed copy)
__global__ void transpose_bf16(const float* __restrict__ W, __bf16* __restrict__ Wt,
                               int R, int Ccols) {
  int i = blockIdx.x * blockDim.x + threadIdx.x;
  if (i >= R * Ccols) return;
  int r = i / Ccols, c = i % Ccols;
  Wt[(long)c * R + r] = f2bf(W[i]);
}

// positional encoding, stored transposed [HW x C] bf16 for the GEMM A operand
__global__ void pe_kernel(__bf16* __restrict__ peT) {
  int i = blockIdx.x * blockDim.x + threadIdx.x;
  if (i >= HWC * CDIM) return;
  int hw = i / CDIM, c = i % CDIM;
  int ii = hw >> 7, jj = hw & 127;
  float di = SC0f * ((float)ii - 63.5f);
  float dj = SC1f * ((float)jj - 63.5f);
  float d = sqrtf(di * di + dj * dj);
  int k = c >> 1;
  float div = __expf(-(float)(2 * k) * 0.0719558032f);  // log(1e4)/128
  float ang = d * div;
  float val = ((c & 1) ? __cosf(ang) : __sinf(ang)) * 0.0883883476f; // 1/sqrt(128)
  peT[i] = f2bf(val);
}

// h[node][c] = x + peproj + conv_b
__global__ void build_h(const float* __restrict__ peproj, const float* __restrict__ conv_b,
                        const float* __restrict__ ego, const float* __restrict__ neb,
                        float* __restrict__ h, __bf16* __restrict__ hbf) {
  long i = (long)blockIdx.x * blockDim.x + threadIdx.x;
  if (i >= (long)NNODE * CDIM) return;
  int node = (int)(i / CDIM), c = (int)(i % CDIM);
  int hw = node & (HWC - 1);
  float x = (node < HWC) ? ego[(long)c * HWC + hw] : neb[(long)c * HWC + hw];
  float v = x + peproj[(long)hw * CDIM + c] + conv_b[c];
  h[i] = v; hbf[i] = f2bf(v);
}

// edge feature init: vals(row) @ edge_w + edge_b
__global__ void edge_init(const float* __restrict__ vr,
                          const float* __restrict__ w1, const float* __restrict__ b1,
                          const float* __restrict__ w2, const float* __restrict__ b2,
                          const float* __restrict__ ew, const float* __restrict__ eb,
                          float* __restrict__ e, __bf16* __restrict__ ebf,
                          int E, int ER, long Epad) {
  long i = (long)blockIdx.x * blockDim.x + threadIdx.x;
  if (i >= Epad * CDIM) return;
  int row = (int)(i / CDIM), c = (int)(i % CDIM);
  float v0, v1, v2, v3;
  if (row >= E) { e[i] = 0.0f; ebf[i] = f2bf(0.0f); return; }
  if (row < 2 * ER) {
    int j = (row < ER) ? row : row - ER;
    float dis = vr[j * 4 + 0], r1 = vr[j * 4 + 1], r2 = vr[j * 4 + 2], r3 = vr[j * 4 + 3];
    float delta = b2[0];
#pragma unroll
    for (int t = 0; t < 8; ++t) delta += (dis * w1[t] + b1[t]) * w2[t];
    float ddd = delta / (dis + 1e-7f);
    float area = SC0f * SC1f;
    float dn = delta * delta / area;
    v0 = (dis + delta) * (1.0f / 3.0f);
    v1 = (r1 + dn) / (1.0f + dn);      // vn1 reduces to the same expression
    v2 = (r2 + ddd) / (1.0f + ddd);
    v3 = (r3 + ddd) / (1.0f + ddd);
    if (row >= ER) v3 = -v3;           // reverse edge
  } else {                              // self loops: [0,1,0,1]
    v0 = 0.0f; v1 = 1.0f; v2 = 0.0f; v3 = 1.0f;
  }
  float val = eb[c] + v0 * ew[0 * CDIM + c] + v1 * ew[1 * CDIM + c]
            + v2 * ew[2 * CDIM + c] + v3 * ew[3 * CDIM + c];
  e[i] = val; ebf[i] = f2bf(val);
}

// ---- attention kernels ----------------------------------------------------

// score[e][c] = K[src]*Q[dst]/4*Ep ; att[e][head] = exp(clip(sum_head, -5, 5))
__global__ __launch_bounds__(256) void edge_score(
    const __bf16* __restrict__ Q, const __bf16* __restrict__ K,
    const __bf16* __restrict__ Ep, const int* __restrict__ src,
    const int* __restrict__ dst, __bf16* __restrict__ score,
    float* __restrict__ att, int E) {
  int t = threadIdx.x;
  int e = blockIdx.x * 2 + (t >> 7);
  int c = t & 127;
  if (e >= E) return;
  int s = src[e], d = dst[e];
  float q  = (float)Q[(long)d * CDIM + c];
  float k  = (float)K[(long)s * CDIM + c];
  float ep = (float)Ep[(long)e * CDIM + c];
  float sc = k * q * 0.25f * ep;   // /sqrt(DH)=4
  score[(long)e * CDIM + c] = f2bf(sc);
  float sum = sc;                  // 16-lane (one head) xor reduction, wave32
#pragma unroll
  for (int m = 1; m < 16; m <<= 1) sum += __shfl_xor(sum, m, 32);
  if ((c & 15) == 0) {
    float cl = fminf(fmaxf(sum, -5.0f), 5.0f);
    att[(long)e * 8 + (c >> 4)] = __expf(cl);
  }
}

__global__ __launch_bounds__(256) void edge_scatter(
    const __bf16* __restrict__ V, const float* __restrict__ att,
    const int* __restrict__ src, const int* __restrict__ dst,
    float* __restrict__ wV, float* __restrict__ z, int E) {
  int t = threadIdx.x;
  int e = blockIdx.x * 2 + (t >> 7);
  int c = t & 127;
  if (e >= E) return;
  int s = src[e], d = dst[e];
  float a = att[(long)e * 8 + (c >> 4)];
  atomicAdd(&wV[(long)d * CDIM + c], a * (float)V[(long)s * CDIM + c]);
  if (c < 8) atomicAdd(&z[(long)d * 8 + c], att[(long)e * 8 + c]);
}

__global__ void hatt_kernel(const float* __restrict__ wV, const float* __restrict__ z,
                            __bf16* __restrict__ hatt) {
  long i = (long)blockIdx.x * blockDim.x + threadIdx.x;
  if (i >= (long)NNODE * CDIM) return;
  int node = (int)(i / CDIM), c = (int)(i % CDIM);
  hatt[i] = f2bf(wV[i] / (z[(long)node * 8 + (c >> 4)] + 1e-6f));
}

// base = LN(base + delta + bias) ; also emit bf16 copy.  one row per block.
__global__ __launch_bounds__(128) void ln_residual(
    float* __restrict__ base, const float* __restrict__ delta,
    const float* __restrict__ bias, __bf16* __restrict__ obf) {
  __shared__ float s1[128], s2[128];
  int row = blockIdx.x, c = threadIdx.x;
  long i = (long)row * CDIM + c;
  float x = base[i] + delta[i] + bias[c];
  s1[c] = x; s2[c] = x * x;
  __syncthreads();
  for (int off = 64; off > 0; off >>= 1) {
    if (c < off) { s1[c] += s1[c + off]; s2[c] += s2[c + off]; }
    __syncthreads();
  }
  float m = s1[0] * (1.0f / 128.0f);
  float var = s2[0] * (1.0f / 128.0f) - m * m;
  float y = (x - m) * rsqrtf(var + 1e-5f);
  base[i] = y; obf[i] = f2bf(y);
}

__global__ void final_out(const float* __restrict__ h, float* __restrict__ out) {
  long i = (long)blockIdx.x * blockDim.x + threadIdx.x;
  if (i >= (long)CDIM * HWC) return;
  int c = (int)(i / HWC), hw = (int)(i % HWC);
  out[i] = h[(long)hw * CDIM + c];
}

// ---------------------------------------------------------------------------

extern "C" void kernel_launch(void* const* d_in, const int* in_sizes, int n_in,
                              void* d_out, int out_size, void* d_ws, size_t ws_size,
                              hipStream_t stream) {
  const float* ego    = (const float*)d_in[0];
  const float* neb    = (const float*)d_in[1];
  const float* vr     = (const float*)d_in[2];
  const float* conv_w = (const float*)d_in[3];
  const float* conv_b = (const float*)d_in[4];
  const float* w1     = (const float*)d_in[5];
  const float* b1     = (const float*)d_in[6];
  const float* w2     = (const float*)d_in[7];
  const float* b2     = (const float*)d_in[8];
  const float* edge_w = (const float*)d_in[9];
  const float* edge_b = (const float*)d_in[10];
  const float* Wq  = (const float*)d_in[11];
  const float* Wk  = (const float*)d_in[12];
  const float* Wv  = (const float*)d_in[13];
  const float* We  = (const float*)d_in[14];
  const float* Wo  = (const float*)d_in[15];
  const float* bo  = (const float*)d_in[16];
  const float* Woe = (const float*)d_in[17];
  const float* boe = (const float*)d_in[18];
  const float* Wf1 = (const float*)d_in[19];
  const float* bf1 = (const float*)d_in[20];
  const float* Wf2 = (const float*)d_in[21];
  const float* bf2 = (const float*)d_in[22];
  const float* Ef1 = (const float*)d_in[23];
  const float* ef1b = (const float*)d_in[24];
  const float* Ef2 = (const float*)d_in[25];
  const float* ef2b = (const float*)d_in[26];
  const int* indices = (const int*)d_in[27];

  const int E  = in_sizes[27] / 2;
  const int ER = in_sizes[2] / 4;
  const long Epad = (long)((E + 127) & ~127);
  const int* srcI = indices;
  const int* dstI = indices + E;

  // ---- workspace arena ----
  char* ws = (char*)d_ws;
  size_t off = 0;
  auto take = [&](size_t bytes) -> void* {
    void* p = ws + off;
    off += (bytes + 255) & ~(size_t)255;
    return p;
  };
  __bf16* convw_bf = (__bf16*)take((size_t)CDIM * CDIM * 2);
  __bf16* wtsL[3];
  for (int l = 0; l < 3; ++l) wtsL[l] = (__bf16*)take((size_t)229376 * 2);
  __bf16* peT   = (__bf16*)take((size_t)HWC * CDIM * 2);
  float*  h     = (float*)take((size_t)NNODE * CDIM * 4);
  __bf16* hbf   = (__bf16*)take((size_t)NNODE * CDIM * 2);
  float*  e     = (float*)take((size_t)Epad * CDIM * 4);
  __bf16* ebf   = (__bf16*)take((size_t)Epad * CDIM * 2);
  __bf16* qbf   = (__bf16*)take((size_t)NNODE * CDIM * 2);           // also h_att
  __bf16* kvbf  = (__bf16*)take((size_t)NNODE * CDIM * 2 * 2);       // K|V, also ffn_h
  __bf16* kbf   = kvbf;
  __bf16* vbf   = kvbf + (size_t)NNODE * CDIM;
  __bf16* epsc  = (__bf16*)take((size_t)Epad * CDIM * 2 * 2);        // Ep|score, also ffn_e
  __bf16* epbf  = epsc;
  __bf16* scbf  = epsc + (size_t)Epad * CDIM;
  float*  att   = (float*)take((size_t)Epad * 8 * 4);
  float*  wV    = (float*)take((size_t)NNODE * CDIM * 4);
  float*  zacc  = (float*)take((size_t)NNODE * 8 * 4);
  float*  tmp_h = (float*)take((size_t)NNODE * CDIM * 4);            // also pe proj
  float*  tmp_e = (float*)take((size_t)Epad * CDIM * 4);

  auto ew = [&](long n) { return dim3((unsigned)((n + 255) / 256)); };
  auto gemm = [&](const __bf16* A, int lda, const __bf16* Bt, int ldb,
                  float* Df, __bf16* Dbf, int ldd, const float* bias,
                  long M, int Ncols, int K, int mode) {
    dim3 g((unsigned)(M / 128), (unsigned)(Ncols / 128));
    wmma_gemm<<<g, 256, 0, stream>>>(A, lda, Bt, ldb, Df, Dbf, ldd, bias, K, mode);
  };

  // ---- weight prep (bf16; second GEMM operand stored [Cout x Cin]) ----
  cvt_bf16<<<ew(CDIM * CDIM), 256, 0, stream>>>(conv_w, convw_bf, CDIM * CDIM);
  for (int l = 0; l < 3; ++l) {
    __bf16* W = wtsL[l];
    const int cc = CDIM * CDIM;
    transpose_bf16<<<ew(cc), 256, 0, stream>>>(Wq  + (size_t)l * cc, W +      0, CDIM, CDIM);
    transpose_bf16<<<ew(cc), 256, 0, stream>>>(Wk  + (size_t)l * cc, W +  16384, CDIM, CDIM);
    transpose_bf16<<<ew(cc), 256, 0, stream>>>(Wv  + (size_t)l * cc, W +  32768, CDIM, CDIM);
    transpose_bf16<<<ew(cc), 256, 0, stream>>>(We  + (size_t)l * cc, W +  49152, CDIM, CDIM);
    transpose_bf16<<<ew(cc), 256, 0, stream>>>(Wo  + (size_t)l * cc, W +  65536, CDIM, CDIM);
    transpose_bf16<<<ew(cc), 256, 0, stream>>>(Woe + (size_t)l * cc, W +  81920, CDIM, CDIM);
    transpose_bf16<<<ew(2 * cc), 256, 0, stream>>>(Wf1 + (size_t)l * 2 * cc, W +  98304, CDIM, 256);
    transpose_bf16<<<ew(2 * cc), 256, 0, stream>>>(Wf2 + (size_t)l * 2 * cc, W + 131072, 256, CDIM);
    transpose_bf16<<<ew(2 * cc), 256, 0, stream>>>(Ef1 + (size_t)l * 2 * cc, W + 163840, CDIM, 256);
    transpose_bf16<<<ew(2 * cc), 256, 0, stream>>>(Ef2 + (size_t)l * 2 * cc, W + 196608, 256, CDIM);
  }

  // ---- node features: h = x + conv(pe) + conv_b ----
  pe_kernel<<<ew((long)HWC * CDIM), 256, 0, stream>>>(peT);
  gemm(peT, CDIM, convw_bf, CDIM, tmp_h, nullptr, CDIM, nullptr, HWC, CDIM, CDIM, 0);
  build_h<<<ew((long)NNODE * CDIM), 256, 0, stream>>>(tmp_h, conv_b, ego, neb, h, hbf);

  // ---- edge features ----
  edge_init<<<ew(Epad * CDIM), 256, 0, stream>>>(vr, w1, b1, w2, b2, edge_w, edge_b,
                                                 e, ebf, E, ER, Epad);

  // ---- 3 transformer layers ----
  for (int l = 0; l < 3; ++l) {
    __bf16* W = wtsL[l];
    // projections (no bias in reference)
    gemm(hbf, CDIM, W +     0, CDIM, nullptr, qbf,  CDIM, nullptr, NNODE, CDIM, CDIM, 2);
    gemm(hbf, CDIM, W + 16384, CDIM, nullptr, kbf,  CDIM, nullptr, NNODE, CDIM, CDIM, 2);
    gemm(hbf, CDIM, W + 32768, CDIM, nullptr, vbf,  CDIM, nullptr, NNODE, CDIM, CDIM, 2);
    gemm(ebf, CDIM, W + 49152, CDIM, nullptr, epbf, CDIM, nullptr, Epad,  CDIM, CDIM, 2);

    // per-edge score + head attention weights
    edge_score<<<dim3((E + 1) / 2), 256, 0, stream>>>(qbf, kbf, epbf, srcI, dstI,
                                                      scbf, att, E);
    // segment sums
    hipMemsetAsync(wV,   0, (size_t)NNODE * CDIM * 4, stream);
    hipMemsetAsync(zacc, 0, (size_t)NNODE * 8 * 4, stream);
    edge_scatter<<<dim3((E + 1) / 2), 256, 0, stream>>>(vbf, att, srcI, dstI, wV, zacc, E);
    hatt_kernel<<<ew((long)NNODE * CDIM), 256, 0, stream>>>(wV, zacc, qbf); // reuse qbf

    // h = LN(h + h_att@Wo + bo)
    gemm(qbf, CDIM, W + 65536, CDIM, tmp_h, nullptr, CDIM, nullptr, NNODE, CDIM, CDIM, 0);
    ln_residual<<<dim3(NNODE), 128, 0, stream>>>(h, tmp_h, bo + (size_t)l * CDIM, hbf);

    // e = LN(e + score@Woe + boe)
    gemm(scbf, CDIM, W + 81920, CDIM, tmp_e, nullptr, CDIM, nullptr, Epad, CDIM, CDIM, 0);
    ln_residual<<<dim3((unsigned)Epad), 128, 0, stream>>>(e, tmp_e, boe + (size_t)l * CDIM, ebf);

    // h FFN: LN(h + relu(h@Wf1+bf1)@Wf2 + bf2); ffn buffer aliases K|V
    gemm(hbf, CDIM, W +  98304, CDIM, nullptr, kvbf, 256, bf1 + (size_t)l * 256, NNODE, 256, CDIM, 1);
    gemm(kvbf, 256, W + 131072, 256, tmp_h, nullptr, CDIM, nullptr, NNODE, CDIM, 256, 0);
    ln_residual<<<dim3(NNODE), 128, 0, stream>>>(h, tmp_h, bf2 + (size_t)l * CDIM, hbf);

    // e FFN; ffn buffer aliases Ep|score
    gemm(ebf, CDIM, W + 163840, CDIM, nullptr, epsc, 256, ef1b + (size_t)l * 256, Epad, 256, CDIM, 1);
    gemm(epsc, 256, W + 196608, 256, tmp_e, nullptr, CDIM, nullptr, Epad, CDIM, 256, 0);
    ln_residual<<<dim3((unsigned)Epad), 128, 0, stream>>>(e, tmp_e, ef2b + (size_t)l * CDIM, ebf);
  }

  // ---- output: h[:HW].T -> (C,H,W) ----
  final_out<<<ew((long)CDIM * HWC), 256, 0, stream>>>(h, (float*)d_out);
}
